// BayesLSTM_5153960755791
// MI455X (gfx1250) — matrix operand order repaired and checked by reading, not compiled, not run on hardware
//
#include <hip/hip_runtime.h>
#include <hip/hip_bf16.h>

// ---------------------------------------------------------------------------
// Bayesian LSTM for MI455X (gfx1250, wave32, WMMA bf16 w/ f32 accumulate).
//
//   * Sample weights once: W = mu + eps*(softplus(rho)+1e-5), stored bf16;
//     KL in the same pass, deterministic 2-pass tree reduction.
//   * Time-parallel half pre-GEMMed: gates0 = X @ Wx + bias (bf16 WMMA).
//   * Recurrence: ONE persistent kernel per layer (16 blocks = CDNA5 cluster
//     limit). Each step: h @ Wh via WMMA, fused LSTM pointwise, then an
//     inter-block barrier (s_cluster_barrier fast path + global-atomic
//     sense-reversing barrier for correctness). Ping-pong bf16 h buffers
//     make one barrier per step sufficient.
//
// Workspace (~337 MB): W bf16 32MB | x bf16 32MB | gates f32 256MB |
//   h/c state f32, 2x h bf16, KL partials, barrier words.
// ---------------------------------------------------------------------------

#define S_LEN 256
#define BATCH 64
#define HID   1024
#define G4    4096   // 4*HID
#define K2    2048   // 2*HID
#define LAYERS 2
#define NBLK_REC 16  // recurrent grid: 16 workgroups (== cluster max)

typedef unsigned int   u32;
typedef unsigned short u16;

typedef __attribute__((ext_vector_type(16))) __bf16 bf16x16;
typedef __attribute__((ext_vector_type(8)))  float  f32x8;

union Frag16 { u32 u[8]; bf16x16 v; };

// ---------------- helpers ----------------

__device__ __forceinline__ u16 f2bf(float f) {
    u32 u = __float_as_uint(f);
    u32 r = (u + 0x7FFFu + ((u >> 16) & 1u)) >> 16;   // round-to-nearest-even
    return (u16)r;
}

__device__ __forceinline__ float sigm(float x) {
    return 1.0f / (1.0f + __expf(-x));
}

// threefry2x32-20 counter-based RNG (deterministic)
__device__ __forceinline__ void threefry2x32(u32 k0, u32 k1, u32 c0, u32 c1,
                                             u32& o0, u32& o1) {
    const u32 ks2 = k0 ^ k1 ^ 0x1BD11BDAu;
    u32 x0 = c0 + k0, x1 = c1 + k1;
    const u32 R[8] = {13u, 15u, 26u, 6u, 17u, 29u, 16u, 24u};
    u32 key[3] = {k0, k1, ks2};
#pragma unroll
    for (int r = 0; r < 20; ++r) {
        u32 rot = R[r & 7];
        x0 += x1;
        x1 = (x1 << rot) | (x1 >> (32u - rot));
        x1 ^= x0;
        if ((r & 3) == 3) {
            u32 s = (u32)((r >> 2) + 1);
            x0 += key[s % 3u];
            x1 += key[(s + 1u) % 3u] + s;
        }
    }
    o0 = x0; o1 = x1;
}

// ---------------- kernel 1: sample weights + KL partials ----------------

#define NW_TOTAL (LAYERS * K2 * G4)        // 16,777,216
#define SAMPLE_BLOCKS 2048

__global__ __launch_bounds__(256)
void sample_weights_kl(const float* __restrict__ mu,
                       const float* __restrict__ rho,
                       u16* __restrict__ wbf,
                       float* __restrict__ partials) {
    const float LOG_2PI    =  1.8378770664093453f;
    const float LOG_PI_MIX = -1.3862943611198906f;   // log 0.25
    const float LOG_1M_PI  = -0.2876820724517809f;   // log 0.75
    const float INV_S1     =  2.7182818284590452f;   // e^1
    const float INV_S2     =  1096.6331584284585f;   // e^7
    const float LOG_S1     = -1.0f;
    const float LOG_S2     = -7.0f;

    const u32 tid    = threadIdx.x;
    const u32 stride = SAMPLE_BLOCKS * 256u;
    float kl_local = 0.0f;

    for (u32 i = blockIdx.x * 256u + tid; i < (u32)NW_TOTAL; i += stride) {
        u32 layer = i >> 23;
        float m = mu[i];
        float r = rho[i];
        float sp = (r > 20.0f) ? r : log1pf(__expf(r));
        float sigma = sp + 1e-5f;

        u32 r0, r1;
        threefry2x32(1u, layer, i, 0x9E3779B9u, r0, r1);
        float u1 = ((float)(r0 >> 8) + 0.5f) * (1.0f / 16777216.0f);
        float u2 = ((float)(r1 >> 8) + 0.5f) * (1.0f / 16777216.0f);
        float rad = sqrtf(-2.0f * __logf(u1));
        float eps = rad * __cosf(6.2831853071795865f * u2);

        float w = m + eps * sigma;
        wbf[i] = f2bf(w);

        float log_q = -0.5f * LOG_2PI - __logf(sigma) - 0.5f * eps * eps;
        float z1 = w * INV_S1;
        float z2 = w * INV_S2;
        float lp1 = -0.5f * LOG_2PI - LOG_S1 - 0.5f * z1 * z1 + LOG_PI_MIX;
        float lp2 = -0.5f * LOG_2PI - LOG_S2 - 0.5f * z2 * z2 + LOG_1M_PI;
        float mx = fmaxf(lp1, lp2);
        float mn = fminf(lp1, lp2);
        float log_p = mx + log1pf(__expf(mn - mx));
        kl_local += (log_q - log_p);
    }

    __shared__ float red[256];
    red[tid] = kl_local;
    __syncthreads();
#pragma unroll
    for (int off = 128; off > 0; off >>= 1) {
        if ((int)tid < off) red[tid] += red[tid + off];
        __syncthreads();
    }
    if (tid == 0) partials[blockIdx.x] = red[0];
}

__global__ __launch_bounds__(256)
void kl_reduce(const float* __restrict__ partials, float* __restrict__ kl_out) {
    __shared__ float red[256];
    float s = 0.0f;
    for (int i = threadIdx.x; i < SAMPLE_BLOCKS; i += 256) s += partials[i];
    red[threadIdx.x] = s;
    __syncthreads();
#pragma unroll
    for (int off = 128; off > 0; off >>= 1) {
        if ((int)threadIdx.x < off) red[threadIdx.x] += red[threadIdx.x + off];
        __syncthreads();
    }
    if (threadIdx.x == 0) kl_out[0] = red[0];
}

// ---------------- kernel 2: x -> bf16 ----------------

__global__ __launch_bounds__(256)
void cvt_x_bf16(const float* __restrict__ x, u16* __restrict__ xbf) {
    int base = (blockIdx.x * 256 + threadIdx.x) * 4;
#pragma unroll
    for (int k = 0; k < 4; ++k) xbf[base + k] = f2bf(x[base + k]);
}

// ---------------- kernel 3: per-layer state init / copy-out ----------------

__global__ __launch_bounds__(256)
void init_state(const float* __restrict__ h0, const float* __restrict__ c0,
                float* __restrict__ hstate, float* __restrict__ cstate,
                u16* __restrict__ hbf0,
                u32* __restrict__ bar_cnt, u32* __restrict__ bar_gen) {
    int i = blockIdx.x * 256 + threadIdx.x;   // 256 blocks -> 65536
    float h = h0[i];
    hstate[i] = h;
    cstate[i] = c0[i];
    hbf0[i] = f2bf(h);
    if (i == 0) { *bar_cnt = 0u; *bar_gen = 0u; }   // reset grid barrier
}

__global__ __launch_bounds__(256)
void copy_state(const float* __restrict__ hstate, const float* __restrict__ cstate,
                float* __restrict__ h_out, float* __restrict__ c_out) {
    int i = blockIdx.x * 256 + threadIdx.x;
    h_out[i] = hstate[i];
    c_out[i] = cstate[i];
}

// ---------------- shared WMMA GEMM tile: 64 rows x 256 cols, K=1024 -------
//
// Block = 256 threads = 8 wave32s. Wave w: m-tile = w>>1 (16 rows),
// n-half = w&1 (8 n-tiles of 16) -> 8 f32 C fragments (64 acc VGPRs).
// colmap 0: contiguous 256-col chunk at n0 (x-GEMM)
// colmap 1: 4 gate quarters x 64-wide hidden slice at n0 (recurrent)

#define APAD 34   // 32 + 2 pad -> 17 dwords/row, coprime with 64 LDS banks

__device__ __forceinline__ void gemm_tile_64x256(
    const u16* __restrict__ A, const u16* __restrict__ W,
    int colmap, int n0, int mrow0,
    u16* smA, u16* smB, f32x8 acc[8]) {

    const int tid   = threadIdx.x;
    const int lane  = tid & 31;
    const int wave  = tid >> 5;
    const int mRow  = (wave >> 1) * 16;
    const int nhalf = wave & 1;

#pragma unroll
    for (int t = 0; t < 8; ++t)
#pragma unroll
        for (int e = 0; e < 8; ++e) acc[t][e] = 0.0f;

    for (int kk = 0; kk < HID; kk += 32) {
        // ---- stage A tile: 64x32 bf16, row-major, padded ----
        for (int u = tid; u < 1024; u += 256) {
            int row = u >> 4;
            int kp  = (u & 15) << 1;
            u32 val = *(const u32*)(A + (size_t)(mrow0 + row) * HID + kk + kp);
            *(u32*)&smA[row * APAD + kp] = val;
        }
        // ---- stage B tile: 32x256 bf16, stored TRANSPOSED [col][k] ----
        for (int u = tid; u < 4096; u += 256) {
            int k  = u >> 7;
            int cc = (u & 127) << 1;
            int col = colmap ? (((cc >> 6) * HID) + n0 + (cc & 63)) : (n0 + cc);
            u32 val = *(const u32*)(W + (size_t)(kk + k) * G4 + col);
            smB[cc * APAD + k]       = (u16)(val & 0xFFFFu);
            smB[(cc + 1) * APAD + k] = (u16)(val >> 16);
        }
        // prefetch next K-slab of W (global_prefetch_b8)
        if (kk + 32 < HID && tid < 32) {
            const u16* p = W + (size_t)(kk + 32 + tid) * G4 + n0;
            __builtin_prefetch((const void*)p, 0, 1);
        }
        __syncthreads();

        // ---- A fragment (16-bit A layout: lanes0-15 K0-7/16-23, +8 hi) ----
        Frag16 a;
        {
            int lh = lane & 15;
            int hi = lane >> 4;
            const u16* base = &smA[(mRow + lh) * APAD];
#pragma unroll
            for (int j = 0; j < 8; ++j) {
                int k = ((j >> 2) << 4) + hi * 8 + ((j & 3) << 1);
                a.u[j] = *(const u32*)(base + k);
            }
        }
        // ---- 8 WMMAs over this wave's n-tiles ----
#pragma unroll
        for (int t = 0; t < 8; ++t) {
            Frag16 b;
            int c     = (nhalf * 8 + t) * 16 + (lane & 15);
            int khalf = (lane >> 4) * 8;
            const u16* bb = &smB[c * APAD + khalf];
#pragma unroll
            for (int j = 0; j < 8; ++j) b.u[j] = *(const u32*)(bb + ((j >> 2) << 4) + ((j & 3) << 1));
            acc[t] = __builtin_amdgcn_wmma_f32_16x16x32_bf16(
                false, a.v, false, b.v, (short)0, acc[t], false, false);
        }
        __syncthreads();
    }
}

// ---------------- kernel 4: x-GEMM (gates0 = X @ Wx + bias) ----------------

__global__ __launch_bounds__(256)
void xgemm_bf16(const u16* __restrict__ A, const u16* __restrict__ W,
                float* __restrict__ gates, const float* __restrict__ bias) {
    __shared__ __align__(16) unsigned char smem[65536];
    u16* smA = (u16*)smem;
    u16* smB = (u16*)(smem + 4352);

    const int lane  = threadIdx.x & 31;
    const int wave  = threadIdx.x >> 5;
    const int mRow  = (wave >> 1) * 16;
    const int nhalf = wave & 1;
    const int mrow0 = blockIdx.y * 64;
    const int n0    = blockIdx.x * 256;

    f32x8 acc[8];
    gemm_tile_64x256(A, W, /*colmap=*/0, n0, mrow0, smA, smB, acc);

#pragma unroll
    for (int t = 0; t < 8; ++t) {
        int nl  = (nhalf * 8 + t) * 16 + (lane & 15);
        int col = n0 + nl;
        float bv = bias[col];
#pragma unroll
        for (int r = 0; r < 8; ++r) {
            int m = mRow + r + ((lane >> 4) << 3);
            gates[(size_t)(mrow0 + m) * G4 + col] = acc[t][r] + bv;
        }
    }
}

// ---------------- inter-block barrier for the persistent recurrence -------
// 16 workgroups. Fast path: CDNA5 cluster barrier (s_barrier_signal/-wait -3)
// -- a NOP unless the dispatch is a 16-WG cluster. Correctness path: global
// atomic sense-reversing barrier (blocks trivially co-resident).

__device__ __forceinline__ void grid_barrier(u32* cnt, u32* gen) {
    __syncthreads();
    __builtin_amdgcn_s_cluster_barrier();   // HW fast path (NOP if unclustered)
    if (threadIdx.x == 0) {
        __threadfence();                    // release h/c writes (agent scope)
        u32 g = __hip_atomic_load(gen, __ATOMIC_ACQUIRE, __HIP_MEMORY_SCOPE_AGENT);
        u32 arrived = __hip_atomic_fetch_add(cnt, 1u, __ATOMIC_ACQ_REL,
                                             __HIP_MEMORY_SCOPE_AGENT);
        if (arrived == (u32)(NBLK_REC - 1)) {
            __hip_atomic_store(cnt, 0u, __ATOMIC_RELAXED, __HIP_MEMORY_SCOPE_AGENT);
            __hip_atomic_fetch_add(gen, 1u, __ATOMIC_RELEASE, __HIP_MEMORY_SCOPE_AGENT);
        } else {
            while (__hip_atomic_load(gen, __ATOMIC_ACQUIRE,
                                     __HIP_MEMORY_SCOPE_AGENT) == g)
                __builtin_amdgcn_s_sleep(1);
        }
        __threadfence();                    // acquire other blocks' h writes
    }
    __syncthreads();
}

// ---------------- kernel 5: persistent recurrent LSTM (one per layer) -----
// Grid = 16 blocks; block owns hidden slice [j0, j0+64) across all 4 gate
// quarters. Per step: WMMA GEMM (h @ Wh), fused pointwise, ping-pong h bf16.

__global__ __launch_bounds__(256)
void lstm_recurrent(const u16* __restrict__ Wh,
                    const float* __restrict__ gates,
                    float* __restrict__ cstate, float* __restrict__ hstate,
                    u16* __restrict__ hbf0, u16* __restrict__ hbf1,
                    float* __restrict__ outp, int writeOut,
                    u32* __restrict__ bar_cnt, u32* __restrict__ bar_gen) {
    __shared__ __align__(16) unsigned char smem[65536];
    u16*   smA = (u16*)smem;
    u16*   smB = (u16*)(smem + 4352);
    float* smG = (float*)smem;              // [64][256] f32, reused post-GEMM

    const int tid   = threadIdx.x;
    const int lane  = tid & 31;
    const int wave  = tid >> 5;
    const int mRow  = (wave >> 1) * 16;
    const int nhalf = wave & 1;
    const int n0    = blockIdx.x * 64;      // hidden slice

#pragma unroll 1
    for (int s = 0; s < S_LEN; ++s) {
        const u16* hin  = (s & 1) ? hbf1 : hbf0;
        u16*       hout = (s & 1) ? hbf0 : hbf1;

        f32x8 acc[8];
        gemm_tile_64x256(hin, Wh, /*colmap=*/1, n0, /*mrow0=*/0, smA, smB, acc);

        // stage C in LDS (cc = quarter*64 + j_local)
#pragma unroll
        for (int t = 0; t < 8; ++t) {
            int nl = (nhalf * 8 + t) * 16 + (lane & 15);
#pragma unroll
            for (int r = 0; r < 8; ++r) {
                int m = mRow + r + ((lane >> 4) << 3);
                smG[m * 256 + nl] = acc[t][r];
            }
        }
        __syncthreads();

        // fused LSTM pointwise over this block's (b, j) elements
        const float* g0 = gates + (size_t)s * BATCH * G4;
        for (int e = tid; e < 64 * 64; e += 256) {
            int b  = e >> 6;
            int jl = e & 63;
            int jg = n0 + jl;
            const float* gr = g0 + (size_t)b * G4;
            float gi = smG[b * 256 +   0 + jl] + gr[          jg];
            float gj = smG[b * 256 +  64 + jl] + gr[HID     + jg];
            float gf = smG[b * 256 + 128 + jl] + gr[2 * HID + jg];
            float go = smG[b * 256 + 192 + jl] + gr[3 * HID + jg];

            int sidx = b * HID + jg;
            float c  = cstate[sidx];
            float nc = c * sigm(gf + 1.0f) + sigm(gi) * tanhf(gj); // FORGET_BIAS
            float nh = tanhf(nc) * sigm(go);

            cstate[sidx] = nc;
            hstate[sidx] = nh;
            hout[sidx]   = f2bf(nh);
            if (writeOut) outp[((size_t)s * BATCH + b) * HID + jg] = nh;
        }

        grid_barrier(bar_cnt, bar_gen);     // h(s+1) visible to all blocks
    }
}

// ---------------- host-side launch sequence ----------------

extern "C" void kernel_launch(void* const* d_in, const int* in_sizes, int n_in,
                              void* d_out, int out_size, void* d_ws, size_t ws_size,
                              hipStream_t stream) {
    const float* x    = (const float*)d_in[0];   // [256,64,1024]
    const float* h0   = (const float*)d_in[1];   // [2,64,1024]
    const float* c0   = (const float*)d_in[2];   // [2,64,1024]
    const float* mu   = (const float*)d_in[3];   // [2,2048,4096]
    const float* rho  = (const float*)d_in[4];   // [2,2048,4096]
    const float* bias = (const float*)d_in[5];   // [2,4096]
    float* out = (float*)d_out;

    const size_t OFF_OUT = 0;
    const size_t OFF_H   = (size_t)S_LEN * BATCH * HID;
    const size_t OFF_C   = OFF_H + (size_t)LAYERS * BATCH * HID;
    const size_t OFF_KL  = OFF_C + (size_t)LAYERS * BATCH * HID;

    char* ws = (char*)d_ws;
    const size_t WS_W     = 0;
    const size_t WS_XBF   = WS_W     + (size_t)NW_TOTAL * 2;
    const size_t WS_GATES = WS_XBF   + (size_t)S_LEN * BATCH * HID * 2;
    const size_t WS_HST   = WS_GATES + (size_t)S_LEN * BATCH * G4 * 4;
    const size_t WS_CST   = WS_HST   + (size_t)BATCH * HID * 4;
    const size_t WS_HBF0  = WS_CST   + (size_t)BATCH * HID * 4;
    const size_t WS_HBF1  = WS_HBF0  + (size_t)BATCH * HID * 2;
    const size_t WS_PART  = WS_HBF1  + (size_t)BATCH * HID * 2;
    const size_t WS_BAR   = WS_PART  + (size_t)SAMPLE_BLOCKS * 4;

    u16*   wbf      = (u16*)(ws + WS_W);
    u16*   xbf      = (u16*)(ws + WS_XBF);
    float* gates    = (float*)(ws + WS_GATES);
    float* hstate   = (float*)(ws + WS_HST);
    float* cstate   = (float*)(ws + WS_CST);
    u16*   hbf0     = (u16*)(ws + WS_HBF0);
    u16*   hbf1     = (u16*)(ws + WS_HBF1);
    float* partials = (float*)(ws + WS_PART);
    u32*   bar_cnt  = (u32*)(ws + WS_BAR);
    u32*   bar_gen  = bar_cnt + 1;

    // 1) weight sampling + KL
    sample_weights_kl<<<SAMPLE_BLOCKS, 256, 0, stream>>>(mu, rho, wbf, partials);
    kl_reduce<<<1, 256, 0, stream>>>(partials, out + OFF_KL);

    // 2) x -> bf16
    cvt_x_bf16<<<16384, 256, 0, stream>>>(x, xbf);

    // 3) layers
    for (int l = 0; l < LAYERS; ++l) {
        const u16* Wl = wbf + (size_t)l * K2 * G4;
        const u16* Wh = Wl + (size_t)HID * G4;
        int writeOut = (l == LAYERS - 1) ? 1 : 0;

        // 3a) gates0 = X @ Wx + bias : M=16384, N=4096, K=1024
        xgemm_bf16<<<dim3(16, 256), 256, 0, stream>>>(
            xbf, Wl, gates, bias + (size_t)l * G4);

        // 3b) init h/c state + barrier words
        init_state<<<256, 256, 0, stream>>>(
            h0 + (size_t)l * BATCH * HID, c0 + (size_t)l * BATCH * HID,
            hstate, cstate, hbf0, bar_cnt, bar_gen);

        // 3c) full recurrence in ONE persistent launch
        lstm_recurrent<<<NBLK_REC, 256, 0, stream>>>(
            Wh, gates, cstate, hstate, hbf0, hbf1,
            out + OFF_OUT, writeOut, bar_cnt, bar_gen);

        // 3d) h_stack / c_stack
        copy_state<<<256, 256, 0, stream>>>(
            hstate, cstate,
            out + OFF_H + (size_t)l * BATCH * HID,
            out + OFF_C + (size_t)l * BATCH * HID);
    }

    (void)in_sizes; (void)n_in; (void)out_size; (void)ws_size;
}
